// LinearAttention_35038343201013
// MI455X (gfx1250) — matrix operand order repaired and checked by reading, not compiled, not run on hardware
//
#include <hip/hip_runtime.h>

typedef __attribute__((ext_vector_type(16))) _Float16 v16h;
typedef __attribute__((ext_vector_type(8)))  _Float16 v8h;
typedef __attribute__((ext_vector_type(4)))  _Float16 v4h;
typedef __attribute__((ext_vector_type(8)))  float    v8f;

#define BQ 4
#define HQ 16
#define SQ 4096
#define DQ 64
#define BH (BQ*HQ)        // 64 heads total
#define SPLIT 16          // S-splits in phase 1
#define ROWS (SQ/SPLIT)   // 256 rows per phase-1 block
#define CHUNK 32          // K-dim chunk per WMMA pass
#define NCH (ROWS/CHUNK)  // 8 chunks
#define TS 64             // S-tile per phase-2 block
#define EPS 1e-6f

// workspace layout (in floats)
#define OFF_CTXPART 0
#define N_CTXPART   (SPLIT*BH*DQ*DQ)            // 4,194,304
#define OFF_NORMPART (OFF_CTXPART + N_CTXPART)
#define N_NORMPART  (SPLIT*BH*DQ)               // 65,536
#define OFF_CTX     (OFF_NORMPART + N_NORMPART)
#define N_CTX       (BH*DQ*DQ)                  // 262,144
#define OFF_NORM    (OFF_CTX + N_CTX)
#define N_NORM      (BH*DQ)                     // 4,096

__device__ __forceinline__ float fmap(float x) {
  // 1 + elu(x): x+1 for x>0, exp(x) for x<=0
  return x > 0.0f ? x + 1.0f : __expf(x);
}

union FragH { v16h v; v8h h[2]; };

// -------- Phase 1: partial context = K_f^T * V, partial normalizer --------
__global__ void __launch_bounds__(256)
la_phase1(const float* __restrict__ keys, const float* __restrict__ values,
          float* __restrict__ ws) {
  __shared__ _Float16 kT[DQ*CHUNK]; // [m][kk]  (A layout: row = d-of-K)
  __shared__ _Float16 vT[DQ*CHUNK]; // [n][kk]  (B^T layout: row = d-of-V)
  __shared__ float nLds[16*DQ];     // per-rowgroup normalizer partials

  const int bh    = blockIdx.y;
  const int split = blockIdx.x;
  const int sBase = split * ROWS;
  const int tid   = threadIdx.x;
  const int lane  = tid & 31, wave = tid >> 5;
  const int hh = lane >> 4, l16 = lane & 15;

  // float4 load ownership: 4 consecutive d-columns, 2 rows per 32-row chunk
  const int d4 = (tid & 15) * 4;
  const int rg = tid >> 4;          // 0..15

  const int tile0 = wave * 2;
  const int tm  = tile0 >> 2;       // M-tile (shared by both output tiles)
  const int tn0 = tile0 & 3;
  const int tn1 = tn0 + 1;

  const size_t base = (size_t)bh * SQ * DQ;

  v8f acc0 = {}; v8f acc1 = {};
  float n0 = 0.f, n1 = 0.f, n2 = 0.f, n3 = 0.f;

  for (int ch = 0; ch < NCH; ++ch) {
    const int s0 = sBase + ch * CHUNK;
#pragma unroll
    for (int half = 0; half < 2; ++half) {
      const int sl = rg + 16 * half;
      const size_t g = base + (size_t)(s0 + sl) * DQ + d4;
      const float4 kv = *(const float4*)&keys[g];
      const float4 vv = *(const float4*)&values[g];
      const float f0 = fmap(kv.x), f1 = fmap(kv.y);
      const float f2 = fmap(kv.z), f3 = fmap(kv.w);
      n0 += f0; n1 += f1; n2 += f2; n3 += f3;
      kT[(d4 + 0) * CHUNK + sl] = (_Float16)f0;
      kT[(d4 + 1) * CHUNK + sl] = (_Float16)f1;
      kT[(d4 + 2) * CHUNK + sl] = (_Float16)f2;
      kT[(d4 + 3) * CHUNK + sl] = (_Float16)f3;
      vT[(d4 + 0) * CHUNK + sl] = (_Float16)vv.x;
      vT[(d4 + 1) * CHUNK + sl] = (_Float16)vv.y;
      vT[(d4 + 2) * CHUNK + sl] = (_Float16)vv.z;
      vT[(d4 + 3) * CHUNK + sl] = (_Float16)vv.w;
    }
    __syncthreads();

    FragH a, b0, b1;
    {
      // A fragment (16x32 f16): lane L -> M = tm*16 + L%16
      const v8h* p = (const v8h*)&kT[(tm * 16 + l16) * CHUNK];
      a.h[0] = p[hh * 2];     // K = hh*16 .. hh*16+7
      a.h[1] = p[hh * 2 + 1]; // K = hh*16+8 .. hh*16+15
      // B fragments (32x16 f16): lane L -> N = tn*16 + L%16, K = hh*16 + i
      const v8h* q0 = (const v8h*)&vT[(tn0 * 16 + l16) * CHUNK];
      b0.h[0] = q0[hh * 2]; b0.h[1] = q0[hh * 2 + 1];
      const v8h* q1 = (const v8h*)&vT[(tn1 * 16 + l16) * CHUNK];
      b1.h[0] = q1[hh * 2]; b1.h[1] = q1[hh * 2 + 1];
    }
    acc0 = __builtin_amdgcn_wmma_f32_16x16x32_f16(false, a.v, false, b0.v,
                                                  (short)0, acc0, false, false);
    acc1 = __builtin_amdgcn_wmma_f32_16x16x32_f16(false, a.v, false, b1.v,
                                                  (short)0, acc1, false, false);
    __syncthreads();
  }

  // store partial context tiles (plain stores -> deterministic reduce later)
  float* ctxPart = ws + OFF_CTXPART + (size_t)(split * BH + bh) * (DQ * DQ);
#pragma unroll
  for (int r = 0; r < 8; ++r) {
    const int m = tm * 16 + r + 8 * hh;  // C/D layout: VGPR r, lanes 16-31 -> M+8
    ctxPart[m * DQ + tn0 * 16 + l16] = acc0[r];
    ctxPart[m * DQ + tn1 * 16 + l16] = acc1[r];
  }

  // deterministic in-block normalizer reduction
  *(float4*)&nLds[rg * DQ + d4] = make_float4(n0, n1, n2, n3);
  __syncthreads();
  if (tid < DQ) {
    float s = 0.f;
#pragma unroll
    for (int g = 0; g < 16; ++g) s += nLds[g * DQ + tid];
    ws[OFF_NORMPART + (split * BH + bh) * DQ + tid] = s;
  }
}

// -------- Reduce: sum split partials into final context & normalizer --------
__global__ void __launch_bounds__(256)
la_reduce(float* __restrict__ ws) {
  const int o = blockIdx.x * 256 + threadIdx.x;
  if (o < N_CTX / 4) {
    const int idx4 = o * 4;
    const int bh = idx4 >> 12;       // / (DQ*DQ)
    const int idx = idx4 & 4095;
    float sx = 0.f, sy = 0.f, sz = 0.f, sw = 0.f;
#pragma unroll
    for (int sp = 0; sp < SPLIT; ++sp) {
      const float4 p =
          *(const float4*)&ws[OFF_CTXPART + (size_t)(sp * BH + bh) * (DQ * DQ) + idx];
      sx += p.x; sy += p.y; sz += p.z; sw += p.w;
    }
    *(float4*)&ws[OFF_CTX + idx4] = make_float4(sx, sy, sz, sw);
  } else {
    const int o2 = o - N_CTX / 4;
    if (o2 < N_NORM) {
      const int bh = o2 >> 6;
      const int dd = o2 & 63;
      float s = 0.f;
#pragma unroll
      for (int sp = 0; sp < SPLIT; ++sp)
        s += ws[OFF_NORMPART + (sp * BH + bh) * DQ + dd];
      ws[OFF_NORM + o2] = s;
    }
  }
}

// -------- Phase 2: out = (Q_f / (Q_f·norm + eps)) * context --------
__global__ void __launch_bounds__(256)
la_phase2(const float* __restrict__ queries, const float* __restrict__ ws,
          float* __restrict__ out) {
  __shared__ _Float16 ctxB[DQ*DQ];  // [e][d] == B^T layout
  __shared__ _Float16 qA[TS*DQ];    // [m][d] == A layout
  __shared__ float fL[TS*68];       // padded f32 features (68 keeps 16B align)
  __shared__ float normL[DQ];
  __shared__ float dPart[TS*4];
  __shared__ float rdenL[TS];

  const int bh  = blockIdx.y;
  const int s0  = blockIdx.x * TS;
  const int tid = threadIdx.x;
  const int lane = tid & 31, wave = tid >> 5;
  const int hh = lane >> 4, l16 = lane & 15;

  // stage context as f16, transposed into B-fragment-friendly layout
  const float* ctxG = ws + OFF_CTX + (size_t)bh * (DQ * DQ);
#pragma unroll
  for (int k = 0; k < 4; ++k) {
    const int idx = (tid + 256 * k) * 4;       // linear over [d][e], e fastest
    const float4 c4 = *(const float4*)&ctxG[idx];
    const int dd = idx >> 6, e = idx & 63;
    ctxB[(e + 0) * DQ + dd] = (_Float16)c4.x;
    ctxB[(e + 1) * DQ + dd] = (_Float16)c4.y;
    ctxB[(e + 2) * DQ + dd] = (_Float16)c4.z;
    ctxB[(e + 3) * DQ + dd] = (_Float16)c4.w;
  }
  if (tid < DQ) normL[tid] = ws[OFF_NORM + bh * DQ + tid];

  // load Q tile (float4), feature map in f32
  const int d4 = (tid & 15) * 4;
  const int rg = tid >> 4;
  const size_t base = (size_t)bh * SQ * DQ;
#pragma unroll
  for (int j = 0; j < 4; ++j) {
    const int m = rg + 16 * j;
    const float4 qv = *(const float4*)&queries[base + (size_t)(s0 + m) * DQ + d4];
    *(float4*)&fL[m * 68 + d4] =
        make_float4(fmap(qv.x), fmap(qv.y), fmap(qv.z), fmap(qv.w));
  }
  __syncthreads();

  // denom = f · norm (fixed-order partial sums -> deterministic)
  {
    const int m = tid & 63, part = tid >> 6;
    float sum = 0.f;
#pragma unroll
    for (int k2 = 0; k2 < 4; ++k2) {
      const int dd = part * 16 + k2 * 4;
      const float4 f = *(const float4*)&fL[m * 68 + dd];
      const float4 nn = *(const float4*)&normL[dd];
      sum += f.x * nn.x + f.y * nn.y + f.z * nn.z + f.w * nn.w;
    }
    dPart[m * 4 + part] = sum;
  }
  __syncthreads();
  if (tid < TS) {
    const float den = dPart[tid * 4] + dPart[tid * 4 + 1] +
                      dPart[tid * 4 + 2] + dPart[tid * 4 + 3] + EPS;
    rdenL[tid] = 1.0f / den;
  }
  __syncthreads();
  // scale + convert to f16 A layout (packed 64-bit LDS stores)
#pragma unroll
  for (int k = 0; k < 4; ++k) {
    const int idx = (tid + 256 * k) * 4;
    const int m = idx >> 6, dd = idx & 63;
    const float4 f = *(const float4*)&fL[m * 68 + dd];
    const float rd = rdenL[m];
    v4h h;
    h[0] = (_Float16)(f.x * rd);
    h[1] = (_Float16)(f.y * rd);
    h[2] = (_Float16)(f.z * rd);
    h[3] = (_Float16)(f.w * rd);
    *(v4h*)&qA[m * DQ + dd] = h;
  }
  __syncthreads();

  // 64x64 = (64x64)*(64x64) GEMM, K looped in two 32-chunks
  const int tile0 = wave * 2;
  const int tm = tile0 >> 2, tn0 = tile0 & 3, tn1 = tn0 + 1;
  v8f acc0 = {}; v8f acc1 = {};
#pragma unroll
  for (int c = 0; c < 2; ++c) {
    FragH a, b0, b1;
    const v8h* p = (const v8h*)&qA[(tm * 16 + l16) * DQ + c * 32];
    a.h[0] = p[hh * 2]; a.h[1] = p[hh * 2 + 1];
    const v8h* q0 = (const v8h*)&ctxB[(tn0 * 16 + l16) * DQ + c * 32];
    b0.h[0] = q0[hh * 2]; b0.h[1] = q0[hh * 2 + 1];
    const v8h* q1 = (const v8h*)&ctxB[(tn1 * 16 + l16) * DQ + c * 32];
    b1.h[0] = q1[hh * 2]; b1.h[1] = q1[hh * 2 + 1];
    acc0 = __builtin_amdgcn_wmma_f32_16x16x32_f16(false, a.v, false, b0.v,
                                                  (short)0, acc0, false, false);
    acc1 = __builtin_amdgcn_wmma_f32_16x16x32_f16(false, a.v, false, b1.v,
                                                  (short)0, acc1, false, false);
  }
#pragma unroll
  for (int r = 0; r < 8; ++r) {
    const int m = tm * 16 + r + 8 * hh;
    const size_t ob = base + (size_t)(s0 + m) * DQ;
    out[ob + tn0 * 16 + l16] = acc0[r];
    out[ob + tn1 * 16 + l16] = acc1[r];
  }
}

extern "C" void kernel_launch(void* const* d_in, const int* in_sizes, int n_in,
                              void* d_out, int out_size, void* d_ws, size_t ws_size,
                              hipStream_t stream) {
  (void)in_sizes; (void)n_in; (void)out_size; (void)ws_size;
  const float* queries = (const float*)d_in[0];
  const float* keys    = (const float*)d_in[1];
  const float* values  = (const float*)d_in[2];
  float* out = (float*)d_out;
  float* ws  = (float*)d_ws;

  la_phase1<<<dim3(SPLIT, BH), 256, 0, stream>>>(keys, values, ws);
  la_reduce<<<dim3((N_CTX / 4 + N_NORM + 255) / 256), 256, 0, stream>>>(ws);
  la_phase2<<<dim3(SQ / TS, BH), 256, 0, stream>>>(queries, ws, out);
}
